// MyTemporalAttention_7584912244778
// MI455X (gfx1250) — matrix operand order repaired and checked by reading, not compiled, not run on hardware
//
#include <hip/hip_runtime.h>
#include <hip/hip_bf16.h>
#include <math.h>

typedef __attribute__((ext_vector_type(16))) _Float16 v16h;
typedef __attribute__((ext_vector_type(8)))  float    v8f;
typedef __attribute__((ext_vector_type(4)))  int      v4i_t;

namespace {
constexpr int kB  = 2;
constexpr int kT  = 128;
constexpr int kN  = 128;
constexpr int kH  = 8;
constexpr int kDh = 64;
constexpr int kF  = 512;                 // FEA
constexpr int kM  = kB * kT * kN;        // 32768 GEMM rows
}

// ---------------------------------------------------------------------------
// CDNA5 async global->LDS path (guarded; falls back to reg staging)
// ---------------------------------------------------------------------------
#if __has_builtin(__builtin_amdgcn_global_load_async_to_lds_b128)
#define ATHENA_ASYNC 1
typedef __attribute__((address_space(1))) v4i_t* gptr_b128;
typedef __attribute__((address_space(3))) v4i_t* lptr_b128;
__device__ __forceinline__ void async_cp_b128(const _Float16* g, _Float16* l) {
  __builtin_amdgcn_global_load_async_to_lds_b128(
      (gptr_b128)g, (lptr_b128)l, 0, 0);
}
#endif

#if __has_builtin(__builtin_amdgcn_s_wait_asynccnt)
#define WAIT_ASYNC0() __builtin_amdgcn_s_wait_asynccnt(0)
#else
#define WAIT_ASYNC0() asm volatile("s_wait_asynccnt 0" ::: "memory")
#endif

// ---------------------------------------------------------------------------
// WMMA helpers (CDNA5 gfx1250, wave32)
// ---------------------------------------------------------------------------
__device__ __forceinline__ v8f wmma_f16(v16h a, v16h b, v8f c) {
  // (neg_a, A, neg_b, B, c_mod, C, reuse_a, reuse_b)
  return __builtin_amdgcn_wmma_f32_16x16x32_f16(false, a, false, b, (short)0, c,
                                                false, false);
}

// A-matrix 16x32 f16 fragment from an LDS tile stored row-major [rows][ld].
// lane&15 -> row M; lanes0-15 hold K 0-15, lanes16-31 hold K 16-31 (matches
// load_frag_b).  One contiguous 32B run -> 2x ds_load_b128.
__device__ __forceinline__ v16h load_frag_a(const _Float16* Ls, int ld,
                                            int m0, int k0, int lane) {
  const int m  = m0 + (lane & 15);
  const int ko = k0 + ((lane >> 4) << 4);
  union { v16h h; uint4 q[2]; } r;
  const _Float16* p = Ls + m * ld + ko;
  r.q[0] = *reinterpret_cast<const uint4*>(p);        // K ko   .. ko+7
  r.q[1] = *reinterpret_cast<const uint4*>(p + 8);    // K ko+8.. ko+15
  return r.h;
}

// B-matrix 32x16 f16 fragment from an LDS tile stored N-major [n][ld] with K
// contiguous. lane&15 -> column N; lanes0-15 hold K 0-15, lanes16-31 K 16-31.
// One contiguous 32B run -> 2x ds_load_b128.
__device__ __forceinline__ v16h load_frag_b(const _Float16* Ls, int ld,
                                            int n0, int k0, int lane) {
  const int n  = n0 + (lane & 15);
  const int ko = k0 + ((lane >> 4) << 4);
  union { v16h h; uint4 q[2]; } r;
  const _Float16* p = Ls + n * ld + ko;
  r.q[0] = *reinterpret_cast<const uint4*>(p);        // K ko  .. ko+7
  r.q[1] = *reinterpret_cast<const uint4*>(p + 8);    // K ko+8.. ko+15
  return r.h;
}

// Same B fragment but gathered from a K-major tile [k][ld] (strided 16-bit
// loads); element j of the fragment is K = ko + j at column n.
__device__ __forceinline__ v16h load_frag_b_strided(const _Float16* Ls, int ld,
                                                    int n0, int k0, int lane) {
  const int n  = n0 + (lane & 15);
  const int ko = k0 + ((lane >> 4) << 4);
  v16h b;
  #pragma unroll
  for (int j = 0; j < 16; ++j) b[j] = Ls[(ko + j) * ld + n];
  return b;
}

// ---------------------------------------------------------------------------
// xt = x + te  (f32 for residual, f16 for WMMA GEMMs)
// ---------------------------------------------------------------------------
__global__ __launch_bounds__(256) void prep_xt_kernel(
    const float* __restrict__ x, const float* __restrict__ te,
    float* __restrict__ xtF, _Float16* __restrict__ xtH) {
  const size_t i = ((size_t)blockIdx.x * 256 + threadIdx.x) * 4;
  const float4 a = *reinterpret_cast<const float4*>(x + i);
  const float4 b = *reinterpret_cast<const float4*>(te + i);
  float4 s;
  s.x = a.x + b.x; s.y = a.y + b.y; s.z = a.z + b.z; s.w = a.w + b.w;
  *reinterpret_cast<float4*>(xtF + i) = s;
  union { _Float16 h[4]; uint2 u; } p;
  p.h[0] = (_Float16)s.x; p.h[1] = (_Float16)s.y;
  p.h[2] = (_Float16)s.z; p.h[3] = (_Float16)s.w;
  *reinterpret_cast<uint2*>(xtH + i) = p.u;
}

// f32 -> f16 weight conversion, TRANSPOSED: WT[n][k] = W[k][n].
__global__ __launch_bounds__(256) void cvt_wT_kernel(
    const float* __restrict__ W, _Float16* __restrict__ WT) {
  const int i = (blockIdx.x * 256 + threadIdx.x) * 4;   // over kF*kF
  const int k = i >> 9;
  const int n = i & (kF - 1);
  const float4 a = *reinterpret_cast<const float4*>(W + (size_t)k * kF + n);
  WT[(size_t)(n + 0) * kF + k] = (_Float16)a.x;
  WT[(size_t)(n + 1) * kF + k] = (_Float16)a.y;
  WT[(size_t)(n + 2) * kF + k] = (_Float16)a.z;
  WT[(size_t)(n + 3) * kF + k] = (_Float16)a.w;
}

// ---------------------------------------------------------------------------
// C[M,512] = A[M,512] @ W[512,512] + bias  (WMMA f16 -> f32 accum)
// Block tile 128x128, 8 waves each owning a 32x64 sub-tile (2x4 WMMA tiles).
// Double-buffered LDS; async global->LDS when available.  WT is [n][k].
// ---------------------------------------------------------------------------
template <bool OUT_H, bool OUT_F, bool RELU>
__global__ __launch_bounds__(256) void gemm512_kernel(
    const _Float16* __restrict__ A, const _Float16* __restrict__ WT,
    const float* __restrict__ bias, _Float16* __restrict__ outH,
    float* __restrict__ outF) {
  __shared__ _Float16 As[2][128 * 32];   // [m][k]
  __shared__ _Float16 Bs[2][128 * 32];   // [n][k]

  const int tid  = threadIdx.x;
  const int lane = tid & 31;
  const int wave = tid >> 5;
  const int bm   = blockIdx.x * 128;
  const int bn   = blockIdx.y * 128;
  const int wm0  = (wave >> 1) * 32;  // wave grid 4x2
  const int wn0  = (wave & 1) * 64;

  // Staging map: 2 threads per row, 16 halves (32B) each, A and B identical.
  const int rS = tid >> 1;
  const int oS = (tid & 1) * 16;
  const _Float16* Arow = A  + (size_t)(bm + rS) * kF + oS;
  const _Float16* Brow = WT + (size_t)(bn + rS) * kF + oS;

  v8f acc[2][4];
  #pragma unroll
  for (int i = 0; i < 2; ++i)
    #pragma unroll
    for (int j = 0; j < 4; ++j) { v8f z = {}; acc[i][j] = z; }

  auto compute = [&](int cur_) {
    const _Float16* Ab = &As[cur_][0];
    const _Float16* Bb = &Bs[cur_][0];
    v16h bf[4];
    #pragma unroll
    for (int j = 0; j < 4; ++j) bf[j] = load_frag_b(Bb, 32, wn0 + j * 16, 0, lane);
    #pragma unroll
    for (int i = 0; i < 2; ++i) {
      const v16h af = load_frag_a(Ab, 32, wm0 + i * 16, 0, lane);
      #pragma unroll
      for (int j = 0; j < 4; ++j) acc[i][j] = wmma_f16(af, bf[j], acc[i][j]);
    }
  };

  constexpr int kSteps = kF / 32;        // 16
  int cur = 0;

#ifdef ATHENA_ASYNC
  // Direct global->LDS DMA (ASYNCcnt), no VGPR round-trip, no ds_store.
  auto stage = [&](int buf, int k0) {
    _Float16* la = &As[buf][rS * 32 + oS];
    _Float16* lb = &Bs[buf][rS * 32 + oS];
    async_cp_b128(Arow + k0, la);
    async_cp_b128(Arow + k0 + 8, la + 8);
    async_cp_b128(Brow + k0, lb);
    async_cp_b128(Brow + k0 + 8, lb + 8);
  };
  stage(0, 0);
  WAIT_ASYNC0();
  __syncthreads();
  for (int step = 0; step < kSteps; ++step) {
    if (step < kSteps - 1) stage(1 - cur, (step + 1) * 32);  // overlap w/ WMMA
    compute(cur);
    WAIT_ASYNC0();
    __syncthreads();
    cur ^= 1;
  }
#else
  uint4 ra0, ra1, rb0, rb1;
  auto ldg = [&](int k0) {
    const uint4* pa = reinterpret_cast<const uint4*>(Arow + k0);
    ra0 = pa[0]; ra1 = pa[1];
    const uint4* pb = reinterpret_cast<const uint4*>(Brow + k0);
    rb0 = pb[0]; rb1 = pb[1];
  };
  auto sts = [&](int buf) {
    uint4* da = reinterpret_cast<uint4*>(&As[buf][rS * 32 + oS]);
    da[0] = ra0; da[1] = ra1;
    uint4* db = reinterpret_cast<uint4*>(&Bs[buf][rS * 32 + oS]);
    db[0] = rb0; db[1] = rb1;
  };
  ldg(0);
  sts(0);
  ldg(32);
  __syncthreads();
  for (int step = 0; step < kSteps; ++step) {
    if (step < kSteps - 1) sts(1 - cur);
    if (step < kSteps - 2) ldg((step + 2) * 32);
    compute(cur);
    __syncthreads();
    cur ^= 1;
  }
#endif

  // Epilogue (branch-free): C layout -> (M = r + 8*(lane>=16), N = lane&15)
  const int hi8  = (lane >> 4) << 3;
  const int ncol = lane & 15;
  #pragma unroll
  for (int j = 0; j < 4; ++j) {
    const int n = bn + wn0 + j * 16 + ncol;
    const float bv = bias[n];
    #pragma unroll
    for (int i = 0; i < 2; ++i)
      #pragma unroll
      for (int r = 0; r < 8; ++r) {
        const int m = bm + wm0 + i * 16 + r + hi8;
        float v = acc[i][j][r] + bv;
        if (RELU) v = fmaxf(v, 0.0f);
        const size_t o = (size_t)m * kF + n;
        if (OUT_H) outH[o] = (_Float16)v;
        if (OUT_F) outF[o] = v;
      }
  }
}

// ---------------------------------------------------------------------------
// Fused attention: one block per (b,h,n). Scores stay in WMMA accumulators;
// masks + softmax in registers; P stored f16 in LDS for the P@V WMMAs.
// Dynamic LDS = 80 KB (Qs/Ks/Vs 16KB each + Ph 32KB).
// ---------------------------------------------------------------------------
__global__ __launch_bounds__(256) void attn_kernel(
    const _Float16* __restrict__ Qg, const _Float16* __restrict__ Kg,
    const _Float16* __restrict__ Vg, _Float16* __restrict__ Og) {
  extern __shared__ _Float16 smem[];
  _Float16* Qs = smem;                // [128][64]
  _Float16* Ks = Qs + 128 * 64;       // [128][64]
  _Float16* Vs = Ks + 128 * 64;       // [128][64]  (natural [t][d])
  _Float16* Ph = Vs + 128 * 64;       // [128][128]

  const int tid  = threadIdx.x;
  const int lane = tid & 31;
  const int wave = tid >> 5;
  const int nn = blockIdx.x % kN;
  const int hh = (blockIdx.x / kN) % kH;
  const int bb = blockIdx.x / (kN * kH);

  {  // stage Q, K, V tiles: 2 threads per sequence row, 32 halves each
    const int s   = tid >> 1;
    const int off = (tid & 1) * 32;
    const size_t rowbase =
        ((size_t)((bb * kT + s) * kN + nn)) * kF + hh * kDh + off;
#ifdef ATHENA_ASYNC
    #pragma unroll
    for (int i = 0; i < 4; ++i) {
      async_cp_b128(Qg + rowbase + i * 8, Qs + s * kDh + off + i * 8);
      async_cp_b128(Kg + rowbase + i * 8, Ks + s * kDh + off + i * 8);
      async_cp_b128(Vg + rowbase + i * 8, Vs + s * kDh + off + i * 8);
    }
    WAIT_ASYNC0();
#else
    const uint4* q4 = reinterpret_cast<const uint4*>(Qg + rowbase);
    const uint4* k4 = reinterpret_cast<const uint4*>(Kg + rowbase);
    const uint4* v4 = reinterpret_cast<const uint4*>(Vg + rowbase);
    uint4* qd = reinterpret_cast<uint4*>(Qs + s * kDh + off);
    uint4* kd = reinterpret_cast<uint4*>(Ks + s * kDh + off);
    uint4* vd = reinterpret_cast<uint4*>(Vs + s * kDh + off);
    #pragma unroll
    for (int i = 0; i < 4; ++i) { qd[i] = q4[i]; kd[i] = k4[i]; vd[i] = v4[i]; }
#endif
  }
  __syncthreads();

  const int s0   = wave * 16;             // this wave's query rows
  const int hi8  = (lane >> 4) << 3;
  const int tcol = lane & 15;

  // ---- S = Q K^T / sqrt(d) ----  (B-frag: n=t from Ks[t][d], K=d contiguous)
  const v16h a0 = load_frag_a(Qs, kDh, s0, 0, lane);
  const v16h a1 = load_frag_a(Qs, kDh, s0, 32, lane);
  v8f S[8];
  #pragma unroll
  for (int j = 0; j < 8; ++j) {
    const v16h b0 = load_frag_b(Ks, kDh, j * 16, 0, lane);
    const v16h b1 = load_frag_b(Ks, kDh, j * 16, 32, lane);
    v8f c = {};
    c = wmma_f16(a0, b0, c);
    c = wmma_f16(a1, b1, c);
    S[j] = c;
  }

  const int seg0 = hh * (kT / kH);
  const int seg1 = (hh == kH - 1) ? kT : seg0 + (kT / kH);
  #pragma unroll
  for (int j = 0; j < 8; ++j) {
    const int t = j * 16 + tcol;
    const bool focus = (t >= seg0) && (t < seg1);
    #pragma unroll
    for (int r = 0; r < 8; ++r) {
      const int s = s0 + r + hi8;
      float v = S[j][r] * 0.125f;           // 1/sqrt(64)
      if (!focus) v *= 0.3f;                // BETA
      if (t > s) v = -__builtin_inff();     // causal
      S[j][r] = v;
    }
  }

  // ---- softmax over t (row spans 16 lanes of one half + 8 tiles) ----
  float mrow[8], srow[8];
  #pragma unroll
  for (int r = 0; r < 8; ++r) {
    float m = -__builtin_inff();
    #pragma unroll
    for (int j = 0; j < 8; ++j) m = fmaxf(m, S[j][r]);
    #pragma unroll
    for (int off = 8; off >= 1; off >>= 1)
      m = fmaxf(m, __shfl_xor(m, off, 32));
    mrow[r] = m;
    srow[r] = 0.0f;
  }
  #pragma unroll
  for (int j = 0; j < 8; ++j)
    #pragma unroll
    for (int r = 0; r < 8; ++r) {
      const float e = __expf(S[j][r] - mrow[r]);
      S[j][r] = e;
      srow[r] += e;
    }
  #pragma unroll
  for (int r = 0; r < 8; ++r) {
    float s = srow[r];
    #pragma unroll
    for (int off = 8; off >= 1; off >>= 1) s += __shfl_xor(s, off, 32);
    srow[r] = 1.0f / s;
  }
  #pragma unroll
  for (int j = 0; j < 8; ++j)
    #pragma unroll
    for (int r = 0; r < 8; ++r)
      Ph[(s0 + r + hi8) * kT + j * 16 + tcol] = (_Float16)(S[j][r] * srow[r]);
  __syncthreads();

  // ---- O = P @ V ----  (B-frag gathered from natural Vs[t][d])
  v16h pa[4];
  #pragma unroll
  for (int kt = 0; kt < 4; ++kt) pa[kt] = load_frag_a(Ph, kT, s0, kt * 32, lane);
  #pragma unroll
  for (int dt = 0; dt < 4; ++dt) {
    v8f c = {};
    #pragma unroll
    for (int kt = 0; kt < 4; ++kt) {
      const v16h bv = load_frag_b_strided(Vs, kDh, dt * 16, kt * 32, lane);
      c = wmma_f16(pa[kt], bv, c);
    }
    #pragma unroll
    for (int r = 0; r < 8; ++r) {
      const int s = s0 + r + hi8;
      const int d = dt * 16 + tcol;
      Og[((size_t)((bb * kT + s) * kN + nn)) * kF + hh * kDh + d] =
          (_Float16)c[r];
    }
  }
}

// ---------------------------------------------------------------------------
// z = LayerNorm(a + res) over last dim 512; one block per row.
// Wave32 shuffle reduction + 8-partial LDS combine (3 barriers total).
// ---------------------------------------------------------------------------
__global__ __launch_bounds__(256) void ln_res_kernel(
    const float* __restrict__ a, const float* __restrict__ res,
    float* __restrict__ outF, _Float16* __restrict__ outH) {
  __shared__ float part[8];
  const size_t row = blockIdx.x;
  const float* pa = a + row * kF;
  const float* pr = res + row * kF;
  const int t    = threadIdx.x;
  const int lane = t & 31;
  const int wv   = t >> 5;

  const float x0 = pa[t] + pr[t];
  const float x1 = pa[t + 256] + pr[t + 256];

  float s = x0 + x1;
  #pragma unroll
  for (int off = 16; off >= 1; off >>= 1) s += __shfl_xor(s, off, 32);
  if (lane == 0) part[wv] = s;
  __syncthreads();
  float tot = 0.0f;
  #pragma unroll
  for (int w = 0; w < 8; ++w) tot += part[w];
  const float mean = tot * (1.0f / (float)kF);
  __syncthreads();

  const float d0 = x0 - mean, d1 = x1 - mean;
  float s2 = d0 * d0 + d1 * d1;
  #pragma unroll
  for (int off = 16; off >= 1; off >>= 1) s2 += __shfl_xor(s2, off, 32);
  if (lane == 0) part[wv] = s2;
  __syncthreads();
  float tot2 = 0.0f;
  #pragma unroll
  for (int w = 0; w < 8; ++w) tot2 += part[w];
  const float inv = rsqrtf(tot2 * (1.0f / (float)kF) + 1e-5f);

  const float y0 = d0 * inv, y1 = d1 * inv;
  outF[row * kF + t] = y0;
  outF[row * kF + t + 256] = y1;
  if (outH) {
    outH[row * kF + t] = (_Float16)y0;
    outH[row * kF + t + 256] = (_Float16)y1;
  }
}

// ---------------------------------------------------------------------------
// Host-side orchestration
// ---------------------------------------------------------------------------
extern "C" void kernel_launch(void* const* d_in, const int* in_sizes, int n_in,
                              void* d_out, int out_size, void* d_ws,
                              size_t ws_size, hipStream_t stream) {
  (void)in_sizes; (void)n_in; (void)out_size; (void)ws_size;
  const float* x  = (const float*)d_in[0];
  const float* te = (const float*)d_in[1];
  const float* Wq = (const float*)d_in[2];  const float* bq = (const float*)d_in[3];
  const float* Wk = (const float*)d_in[4];  const float* bk = (const float*)d_in[5];
  const float* Wv = (const float*)d_in[6];  const float* bv = (const float*)d_in[7];
  const float* Wo = (const float*)d_in[8];  const float* bo = (const float*)d_in[9];
  const float* W1 = (const float*)d_in[10]; const float* b1 = (const float*)d_in[11];
  const float* W2 = (const float*)d_in[12]; const float* b2 = (const float*)d_in[13];
  float* out = (float*)d_out;

  // Workspace layout (~259 MB), with aliasing once producers are consumed.
  char* ws = (char*)d_ws;
  const size_t MF = (size_t)kM * kF;               // 16,777,216 elements
  float*    xtF   = (float*)(ws);                  // 64 MB  (later: z, in-place)
  _Float16* xtH   = (_Float16*)(ws + MF * 4);      // 32 MB  (later: attn output)
  _Float16* qHbuf = (_Float16*)(ws + MF * 6);      // 32 MB  (later: h1)
  _Float16* kHbuf = (_Float16*)(ws + MF * 8);      // 32 MB  (later: z f16)
  _Float16* vHbuf = (_Float16*)(ws + MF * 10);     // 32 MB
  float*    oF    = (float*)(ws + MF * 12);        // 64 MB  (later: h2)
  _Float16* Wh    = (_Float16*)(ws + MF * 16);     // 6 x 512 KB (transposed)
  const size_t WSZ = (size_t)kF * kF;
  _Float16* WqT = Wh + 0 * WSZ; _Float16* WkT = Wh + 1 * WSZ;
  _Float16* WvT = Wh + 2 * WSZ; _Float16* WoT = Wh + 3 * WSZ;
  _Float16* W1T = Wh + 4 * WSZ; _Float16* W2T = Wh + 5 * WSZ;

  _Float16* attnH = xtH;    // xt f16 dead after QKV GEMMs
  float*    zF    = xtF;    // LN writes row-in-place (read-all-then-write)
  _Float16* zH    = kHbuf;  // kHbuf dead after attention
  _Float16* h1H   = qHbuf;  // qHbuf dead after attention
  float*    h2F   = oF;     // oF dead after first residual-LN

  const dim3 blk(256);
  const dim3 gGemm(kM / 128, kF / 128);             // 256 x 4
  const int  gW   = (kF * kF) / (256 * 4);          // 256
  const int  gXt  = (int)(MF / (256 * 4));          // 16384
  const int  gAtt = kB * kH * kN;                   // 2048
  const unsigned attnLds =
      (3 * 128 * 64 + 128 * 128) * sizeof(_Float16);  // 80 KB

  cvt_wT_kernel<<<gW, blk, 0, stream>>>(Wq, WqT);
  cvt_wT_kernel<<<gW, blk, 0, stream>>>(Wk, WkT);
  cvt_wT_kernel<<<gW, blk, 0, stream>>>(Wv, WvT);
  cvt_wT_kernel<<<gW, blk, 0, stream>>>(Wo, WoT);
  cvt_wT_kernel<<<gW, blk, 0, stream>>>(W1, W1T);
  cvt_wT_kernel<<<gW, blk, 0, stream>>>(W2, W2T);

  prep_xt_kernel<<<gXt, blk, 0, stream>>>(x, te, xtF, xtH);

  gemm512_kernel<true, false, false><<<gGemm, blk, 0, stream>>>(
      xtH, WqT, bq, qHbuf, nullptr);
  gemm512_kernel<true, false, false><<<gGemm, blk, 0, stream>>>(
      xtH, WkT, bk, kHbuf, nullptr);
  gemm512_kernel<true, false, false><<<gGemm, blk, 0, stream>>>(
      xtH, WvT, bv, vHbuf, nullptr);

  attn_kernel<<<gAtt, blk, attnLds, stream>>>(qHbuf, kHbuf, vHbuf, attnH);

  gemm512_kernel<false, true, false><<<gGemm, blk, 0, stream>>>(
      attnH, WoT, bo, nullptr, oF);
  ln_res_kernel<<<kM, blk, 0, stream>>>(oF, xtF, zF, zH);

  gemm512_kernel<true, false, true><<<gGemm, blk, 0, stream>>>(
      zH, W1T, b1, h1H, nullptr);
  gemm512_kernel<false, true, false><<<gGemm, blk, 0, stream>>>(
      h1H, W2T, b2, nullptr, h2F);
  ln_res_kernel<<<kM, blk, 0, stream>>>(h2F, zF, out, nullptr);
}
